// CrossModalAttention_69655779606753
// MI455X (gfx1250) — matrix operand order repaired and checked by reading, not compiled, not run on hardware
//
#include <hip/hip_runtime.h>
#include <stdint.h>

// ---------------- types ----------------
typedef __bf16 bf16_t;
typedef __attribute__((ext_vector_type(16))) __bf16 v16bf;
typedef __attribute__((ext_vector_type(8)))  __bf16 v8bf;
typedef __attribute__((ext_vector_type(8)))  float  v8f;
typedef __attribute__((ext_vector_type(4)))  int    v4i;

// ---------------- problem constants ----------------
constexpr int kB       = 8;
constexpr int kSeq     = 1024;
constexpr int kDim     = 768;
constexpr int kHeads   = 12;
constexpr int kHd      = 64;              // head dim
constexpr int kRows    = kB * kSeq;       // 8192
constexpr int kQKVCols = 3 * kDim;        // 2304
constexpr float kScale = 0.125f;          // 64^-0.5

// ---------------- CDNA5 async LDS copy (ASYNCcnt path) ----------------
#if defined(__gfx1250__) && __has_builtin(__builtin_amdgcn_global_load_async_to_lds_b128)
#define ATTN_ASYNC_LDS 1
typedef __attribute__((address_space(1))) v4i* gv4i_p;   // global (AS1)
typedef __attribute__((address_space(3))) v4i* lv4i_p;   // LDS (AS3)
__device__ inline void async_cp16(const bf16_t* g, bf16_t* l) {
  // global -> LDS, 16B per lane, tracked by ASYNCcnt (no VGPR data round-trip)
  __builtin_amdgcn_global_load_async_to_lds_b128(
      (gv4i_p)(uintptr_t)g,
      (lv4i_p)(uint32_t)(uintptr_t)l,
      /*offset=*/0, /*cpol=*/0);
}
__device__ inline void wait_async0() {
#if __has_builtin(__builtin_amdgcn_s_wait_asynccnt)
  __builtin_amdgcn_s_wait_asynccnt(0);
#else
  asm volatile("s_wait_asynccnt 0x0" ::: "memory");
#endif
}
#else
#define ATTN_ASYNC_LDS 0
#endif

// ---------------- helpers ----------------
__device__ inline v8f wmma_bf16(v16bf a, v16bf b, v8f c) {
  // D = A(16x32 bf16) * B(32x16 bf16) + C(16x16 f32)
  return __builtin_amdgcn_wmma_f32_16x16x32_bf16(
      /*neg_a=*/false, a, /*neg_b=*/false, b,
      /*c_mod=*/(short)0, c, /*reuse_a=*/false, /*reuse_b=*/false);
}

__device__ inline v8f zero_v8f() {
  v8f z;
#pragma unroll
  for (int r = 0; r < 8; ++r) z[r] = 0.0f;
  return z;
}

// wave32 xor-shuffle via ds_swizzle group-of-32 mode: imm = (xor<<10)|0x1f
template <int IMM>
__device__ inline void redmax8(float t[8]) {
#pragma unroll
  for (int r = 0; r < 8; ++r)
    t[r] = fmaxf(t[r], __int_as_float(
        __builtin_amdgcn_ds_swizzle(__float_as_int(t[r]), IMM)));
}
template <int IMM>
__device__ inline void redsum8(float t[8]) {
#pragma unroll
  for (int r = 0; r < 8; ++r)
    t[r] += __int_as_float(
        __builtin_amdgcn_ds_swizzle(__float_as_int(t[r]), IMM));
}

// A fragment (16x32 bf16, row-major source, row stride `ld` elements).
// Lane mapping (ISA 7.12.2): row = lane%16; elems 0..7 -> k = half*8..,
// elems 8..15 -> k = 16 + half*8.. (two contiguous 16B chunks).
__device__ inline v16bf load_afrag(const bf16_t* base, int ld, int ln, int half) {
  const bf16_t* p = base + (size_t)ln * ld + half * 8;
  v16bf a;
  ((v8bf*)&a)[0] = *(const v8bf*)(p);
  ((v8bf*)&a)[1] = *(const v8bf*)(p + 16);
  return a;
}

// B fragment (32x16 bf16) from Bt stored as [n][k] row-major (i.e. B^T).
// Lane mapping: col n = lane%16; elems 0..7 -> k = half*8.., elems 8..15 -> k = 16 + half*8..
__device__ inline v16bf load_bfrag(const bf16_t* baseT, int ld, int ln, int half) {
  const bf16_t* p = baseT + (size_t)ln * ld + half * 8;
  v16bf b;
  ((v8bf*)&b)[0] = *(const v8bf*)(p);
  ((v8bf*)&b)[1] = *(const v8bf*)(p + 16);
  return b;
}

// ---------------- conversion kernels ----------------
__global__ void k_cvt(const float* __restrict__ s, bf16_t* __restrict__ d, int n) {
  int i = blockIdx.x * blockDim.x + threadIdx.x;
  if (i < n) d[i] = (bf16_t)s[i];
}

// dst[c*R + r] = (bf16) src[r*C + c]   (store W transposed for B-frag loads)
__global__ void k_cvt_t(const float* __restrict__ s, bf16_t* __restrict__ d,
                        int R, int C) {
  int i = blockIdx.x * blockDim.x + threadIdx.x;
  if (i < R * C) {
    int r = i / C, c = i - r * C;
    d[(size_t)c * R + r] = (bf16_t)s[i];
  }
}

// ---------------- GEMM: C[M,N] = A[M,K](bf16) * Bt[N,K]^T + bias ----------------
// EPI==0: scatter into Q(*scale) [bh][n][d], K [bh][n][d], Vt [bh][d][n] (bf16)
// EPI==1: write f32 row-major
template <int EPI>
__global__ __launch_bounds__(128) void k_gemm(
    const bf16_t* __restrict__ A, const bf16_t* __restrict__ Bt,
    const float* __restrict__ bias, float* __restrict__ Cf,
    bf16_t* __restrict__ Qb, bf16_t* __restrict__ Kb, bf16_t* __restrict__ Vt,
    int M, int N, int K) {
  const int lane = threadIdx.x & 31;
  const int w    = threadIdx.x >> 5;   // 4 waves, 2x2 of 64x64 tiles
  const int half = lane >> 4;
  const int ln   = lane & 15;
  const int m0 = blockIdx.x * 128 + (w >> 1) * 64;
  const int n0 = blockIdx.y * 128 + (w & 1) * 64;

  v8f acc[4][4];
#pragma unroll
  for (int i = 0; i < 4; ++i)
#pragma unroll
    for (int j = 0; j < 4; ++j) acc[i][j] = zero_v8f();

  for (int kk = 0; kk < K; kk += 32) {
    v16bf af[4], bfr[4];
#pragma unroll
    for (int i = 0; i < 4; ++i)
      af[i] = load_afrag(A + (size_t)(m0 + i * 16) * K + kk, K, ln, half);
#pragma unroll
    for (int j = 0; j < 4; ++j)
      bfr[j] = load_bfrag(Bt + (size_t)(n0 + j * 16) * K + kk, K, ln, half);
#pragma unroll
    for (int i = 0; i < 4; ++i)
#pragma unroll
      for (int j = 0; j < 4; ++j)
        acc[i][j] = wmma_bf16(af[i], bfr[j], acc[i][j]);
  }

  if (EPI == 1) {
#pragma unroll
    for (int j = 0; j < 4; ++j) {
      const int n = n0 + j * 16 + ln;
      const float bv = bias[n];
#pragma unroll
      for (int i = 0; i < 4; ++i)
#pragma unroll
        for (int r = 0; r < 8; ++r)
          Cf[(size_t)(m0 + i * 16 + r + 8 * half) * N + n] = acc[i][j][r] + bv;
    }
  } else {
#pragma unroll
    for (int j = 0; j < 4; ++j) {
      // sel/h are uniform over the 16 columns of a tile (768, 64 | 16)
      const int nt  = n0 + j * 16;
      const int sel = nt / kDim;               // 0=q 1=k 2=v
      const int c0  = nt - sel * kDim;
      const int h   = c0 >> 6;
      const int dd  = (c0 & 63) + ln;
      const float bv = bias[nt + ln];
      if (sel == 0) {
#pragma unroll
        for (int i = 0; i < 4; ++i)
#pragma unroll
          for (int r = 0; r < 8; ++r) {
            const int m = m0 + i * 16 + r + 8 * half;
            const size_t bh = (size_t)((m >> 10) * kHeads + h);
            Qb[(bh * kSeq + (m & 1023)) * kHd + dd] =
                (bf16_t)((acc[i][j][r] + bv) * kScale);  // fold softmax scale
          }
      } else if (sel == 1) {
#pragma unroll
        for (int i = 0; i < 4; ++i)
#pragma unroll
          for (int r = 0; r < 8; ++r) {
            const int m = m0 + i * 16 + r + 8 * half;
            const size_t bh = (size_t)((m >> 10) * kHeads + h);
            Kb[(bh * kSeq + (m & 1023)) * kHd + dd] = (bf16_t)(acc[i][j][r] + bv);
          }
      } else {
#pragma unroll
        for (int i = 0; i < 4; ++i)
#pragma unroll
          for (int r = 0; r < 8; ++r) {
            const int m = m0 + i * 16 + r + 8 * half;
            const size_t bh = (size_t)((m >> 10) * kHeads + h);
            Vt[(bh * kHd + dd) * kSeq + (m & 1023)] = (bf16_t)(acc[i][j][r] + bv);
          }
      }
    }
  }
}

// ---------------- flash attention ----------------
// grid (96, 8), block 256 (8 waves). Wave handles 16 queries; block shares
// LDS-staged 64-key K/V tiles. With async-LDS: double-buffered tiles filled
// by GLOBAL_LOAD_ASYNC_TO_LDS_B128 fully overlapped with WMMA + softmax.
__global__ __launch_bounds__(256) void k_attn(
    const bf16_t* __restrict__ Qb, const bf16_t* __restrict__ Kb,
    const bf16_t* __restrict__ Vt, bf16_t* __restrict__ Oc) {
#if ATTN_ASYNC_LDS
  __shared__ __align__(16) bf16_t kt_lds[2][64 * 64];   // [key][d]   16 KB
  __shared__ __align__(16) bf16_t vt_lds[2][64 * 64];   // [d][key]   16 KB
#else
  __shared__ __align__(16) bf16_t kt_lds[1][64 * 64];   // [key][d]    8 KB
  __shared__ __align__(16) bf16_t vt_lds[1][64 * 64];   // [d][key]    8 KB
#endif
  __shared__ __align__(16) bf16_t pstage[8][16 * 64];   // per-wave P 16 KB

  const int tid  = threadIdx.x;
  const int lane = tid & 31;
  const int w    = tid >> 5;
  const int half = lane >> 4;
  const int ln   = lane & 15;
  const int bh = blockIdx.x;
  const int bb = bh / kHeads;
  const int h  = bh - bb * kHeads;
  const int q0 = blockIdx.y * 128 + w * 16;

  const bf16_t* Qh = Qb + (size_t)bh * kSeq * kHd;
  const bf16_t* Kh = Kb + (size_t)bh * kSeq * kHd;
  const bf16_t* Vh = Vt + (size_t)bh * kHd * kSeq;

  // cooperative-fill assignment: 256 threads x 32B cover one 64x64 bf16 tile
  const int frow = tid >> 2;          // 0..63
  const int fcol = (tid & 3) * 16;    // 0,16,32,48

  // resident Q fragments (16x64 = two 16x32 A-frags), already scaled by 1/8
  v16bf aq[2];
#pragma unroll
  for (int s = 0; s < 2; ++s)
    aq[s] = load_afrag(Qh + (size_t)q0 * kHd + s * 32, kHd, ln, half);

  float mrow[8], lrow[8];
  v8f o[4];
#pragma unroll
  for (int r = 0; r < 8; ++r) { mrow[r] = -3.0e38f; lrow[r] = 0.0f; }
#pragma unroll
  for (int j = 0; j < 4; ++j) o[j] = zero_v8f();

#if ATTN_ASYNC_LDS
  // ---- issue tile 0 copies (ASYNCcnt) ----
  {
    const bf16_t* kp = Kh + (size_t)frow * kHd + fcol;
    async_cp16(kp,     &kt_lds[0][frow * 64 + fcol]);
    async_cp16(kp + 8, &kt_lds[0][frow * 64 + fcol + 8]);
    const bf16_t* vp = Vh + (size_t)frow * kSeq + fcol;
    async_cp16(vp,     &vt_lds[0][frow * 64 + fcol]);
    async_cp16(vp + 8, &vt_lds[0][frow * 64 + fcol + 8]);
  }
#else
  // ---- prefetch tile 0 into registers ----
  v8bf kr0, kr1, vr0, vr1;
  {
    const bf16_t* kp = Kh + (size_t)frow * kHd + fcol;
    kr0 = *(const v8bf*)kp;  kr1 = *(const v8bf*)(kp + 8);
    const bf16_t* vp = Vh + (size_t)frow * kSeq + fcol;
    vr0 = *(const v8bf*)vp;  vr1 = *(const v8bf*)(vp + 8);
  }
#endif

  for (int kt = 0; kt < kSeq / 64; ++kt) {
#if ATTN_ASYNC_LDS
    const int cur = kt & 1;
    wait_async0();          // this wave's copies into buffer `cur` done
    __syncthreads();        // ...and everyone else's
    if (kt + 1 < kSeq / 64) {   // kick next tile into the other buffer
      const int k0n = (kt + 1) * 64, nxt = cur ^ 1;
      const bf16_t* kp = Kh + (size_t)(k0n + frow) * kHd + fcol;
      async_cp16(kp,     &kt_lds[nxt][frow * 64 + fcol]);
      async_cp16(kp + 8, &kt_lds[nxt][frow * 64 + fcol + 8]);
      const bf16_t* vp = Vh + (size_t)frow * kSeq + k0n + fcol;
      async_cp16(vp,     &vt_lds[nxt][frow * 64 + fcol]);
      async_cp16(vp + 8, &vt_lds[nxt][frow * 64 + fcol + 8]);
    }
    const bf16_t* kbuf = &kt_lds[cur][0];
    const bf16_t* vbuf = &vt_lds[cur][0];
#else
    // ---- stage prefetched K/V tile into LDS ----
    *(v8bf*)&kt_lds[0][frow * 64 + fcol]     = kr0;
    *(v8bf*)&kt_lds[0][frow * 64 + fcol + 8] = kr1;
    *(v8bf*)&vt_lds[0][frow * 64 + fcol]     = vr0;
    *(v8bf*)&vt_lds[0][frow * 64 + fcol + 8] = vr1;
    __syncthreads();
    if (kt + 1 < kSeq / 64) {   // next tile's global loads overlap compute
      const int k0n = (kt + 1) * 64;
      const bf16_t* kp = Kh + (size_t)(k0n + frow) * kHd + fcol;
      kr0 = *(const v8bf*)kp;  kr1 = *(const v8bf*)(kp + 8);
      const bf16_t* vp = Vh + (size_t)frow * kSeq + k0n + fcol;
      vr0 = *(const v8bf*)vp;  vr1 = *(const v8bf*)(vp + 8);
    }
    const bf16_t* kbuf = &kt_lds[0][0];
    const bf16_t* vbuf = &vt_lds[0][0];
#endif

    // ---- S = Q * K^T : four 16x16 tiles over 64 keys ----
    v8f s[4];
#pragma unroll
    for (int nj = 0; nj < 4; ++nj) {
      s[nj] = zero_v8f();
#pragma unroll
      for (int ks = 0; ks < 2; ++ks) {
        v16bf bk = load_bfrag(kbuf + (nj * 16) * 64 + ks * 32, 64, ln, half);
        s[nj] = wmma_bf16(aq[ks], bk, s[nj]);
      }
    }

    // ---- online softmax (row = r+8*half lives in a 16-lane half-group) ----
    float t[8];
#pragma unroll
    for (int r = 0; r < 8; ++r)
      t[r] = fmaxf(fmaxf(s[0][r], s[1][r]), fmaxf(s[2][r], s[3][r]));
    redmax8<0x041f>(t); redmax8<0x081f>(t);
    redmax8<0x101f>(t); redmax8<0x201f>(t);
    float corr[8];
#pragma unroll
    for (int r = 0; r < 8; ++r) {
      const float mnew = fmaxf(mrow[r], t[r]);
      corr[r] = __expf(mrow[r] - mnew);
      mrow[r] = mnew;
#pragma unroll
      for (int nj = 0; nj < 4; ++nj) s[nj][r] = __expf(s[nj][r] - mnew);
      t[r] = (s[0][r] + s[1][r]) + (s[2][r] + s[3][r]);
    }
    redsum8<0x041f>(t); redsum8<0x081f>(t);
    redsum8<0x101f>(t); redsum8<0x201f>(t);
#pragma unroll
    for (int r = 0; r < 8; ++r) lrow[r] = lrow[r] * corr[r] + t[r];
#pragma unroll
    for (int j = 0; j < 4; ++j)
#pragma unroll
      for (int r = 0; r < 8; ++r) o[j][r] *= corr[r];

    // ---- relayout P: C-layout -> 16x64 row-major in LDS -> two A-frags ----
#pragma unroll
    for (int nj = 0; nj < 4; ++nj)
#pragma unroll
      for (int r = 0; r < 8; ++r)
        pstage[w][(r + 8 * half) * 64 + nj * 16 + ln] = (bf16_t)s[nj][r];
    __syncthreads();
    v16bf ap[2];
    ap[0] = load_afrag(&pstage[w][0],  64, ln, half);
    ap[1] = load_afrag(&pstage[w][32], 64, ln, half);

    // ---- O += P * V  (vbuf is [d][key] so B-frags are contiguous) ----
#pragma unroll
    for (int j = 0; j < 4; ++j)
#pragma unroll
      for (int ks = 0; ks < 2; ++ks) {
        v16bf bv = load_bfrag(vbuf + (j * 16) * 64 + ks * 32, 64, ln, half);
        o[j] = wmma_bf16(ap[ks], bv, o[j]);
      }
#if !ATTN_ASYNC_LDS
    __syncthreads();   // single-buffer: protect kt/vt_lds before next store
#endif
  }

  // ---- normalize + store to Oc[b*1024+q][h*64+d] (bf16, GEMM2 A-layout) ----
#pragma unroll
  for (int r = 0; r < 8; ++r) {
    const float inv = 1.0f / lrow[r];
    const int q = q0 + r + 8 * half;
    const size_t row = (size_t)bb * kSeq + q;
#pragma unroll
    for (int j = 0; j < 4; ++j)
      Oc[row * kDim + h * kHd + j * 16 + ln] = (bf16_t)(o[j][r] * inv);
  }
}

// ---------------- host launcher ----------------
extern "C" void kernel_launch(void* const* d_in, const int* in_sizes, int n_in,
                              void* d_out, int out_size, void* d_ws, size_t ws_size,
                              hipStream_t stream) {
  const float* x      = (const float*)d_in[0];
  // d_in[1] = context: intentionally unused (reference ignores it)
  const float* qkv_w  = (const float*)d_in[2];
  const float* qkv_b  = (const float*)d_in[3];
  const float* proj_w = (const float*)d_in[4];
  const float* proj_b = (const float*)d_in[5];
  float* out = (float*)d_out;

  // workspace carve-up (bf16 buffers)
  char* ws = (char*)d_ws;
  const size_t szXb  = (size_t)kRows * kDim * 2;       // 12.6 MB
  const size_t szWq  = (size_t)kDim * kQKVCols * 2;    //  3.5 MB
  const size_t szWp  = (size_t)kDim * kDim * 2;        //  1.2 MB
  const size_t szHd  = (size_t)kB * kHeads * kSeq * kHd * 2;  // 12.6 MB each
  bf16_t* Xb   = (bf16_t*)(ws);
  bf16_t* Wqkv = (bf16_t*)(ws + szXb);
  bf16_t* Wprj = (bf16_t*)(ws + szXb + szWq);
  bf16_t* Qb   = (bf16_t*)(ws + szXb + szWq + szWp);
  bf16_t* Kb   = (bf16_t*)(ws + szXb + szWq + szWp + szHd);
  bf16_t* Vt   = (bf16_t*)(ws + szXb + szWq + szWp + 2 * szHd);
  bf16_t* Oc   = (bf16_t*)(ws + szXb + szWq + szWp + 3 * szHd);

  const int nx = kRows * kDim;
  k_cvt<<<(nx + 255) / 256, 256, 0, stream>>>(x, Xb, nx);
  k_cvt_t<<<(kDim * kQKVCols + 255) / 256, 256, 0, stream>>>(qkv_w, Wqkv, kDim, kQKVCols);
  k_cvt_t<<<(kDim * kDim + 255) / 256, 256, 0, stream>>>(proj_w, Wprj, kDim, kDim);

  // QKV projection: [8192 x 2304] = Xb @ Wqkv^T, scatter to Q(*scale)/K/Vt
  k_gemm<0><<<dim3(kRows / 128, kQKVCols / 128), 128, 0, stream>>>(
      Xb, Wqkv, qkv_b, nullptr, Qb, Kb, Vt, kRows, kQKVCols, kDim);

  // flash attention per (b,h), 128 queries per block
  k_attn<<<dim3(kB * kHeads, kSeq / 128), 256, 0, stream>>>(Qb, Kb, Vt, Oc);

  // output projection: [8192 x 768] f32 = Oc @ Wprj^T
  k_gemm<1><<<dim3(kRows / 128, kDim / 128), 128, 0, stream>>>(
      Oc, Wprj, proj_b, out, nullptr, nullptr, nullptr, kRows, kDim, kDim);
}
